// ScaledDotProductAttention_80066780332533
// MI455X (gfx1250) — matrix operand order repaired and checked by reading, not compiled, not run on hardware
//
#include <hip/hip_runtime.h>

typedef __attribute__((ext_vector_type(16))) __bf16 v16bf;
typedef __attribute__((ext_vector_type(8)))  __bf16 v8bf;
typedef __attribute__((ext_vector_type(2)))  __bf16 v2bf;
typedef __attribute__((ext_vector_type(8)))  float  v8f;

#define BATCH 16
#define SEQ   2048
#define DIM   256
#define QROWS 128       // q rows per block (8 waves x 16)
#define NWAVE 8
#define THREADS 256
#define KTILE 32        // kv rows per k-step (== WMMA K for the PV product)
#define NT    (SEQ / KTILE)
#define QSTR  264       // padded LDS row strides (bank-conflict-free frag reads)
#define KSTR  264
#define VSTR  40
#define PSTR  40

// CDNA5 async DMA: global -> LDS, 16B per lane, tracked by ASYNCcnt.
__device__ __forceinline__ void async_ld_b128(const void* g, unsigned lds_off) {
  unsigned long long ga = (unsigned long long)g;
  asm volatile("global_load_async_to_lds_b128 %0, %1, off"
               :: "v"(lds_off), "v"(ga) : "memory");
}
__device__ __forceinline__ void wait_async0() {
  asm volatile("s_wait_asynccnt 0" ::: "memory");
}

__device__ __forceinline__ v16bf ld_frag2(const __bf16* p0, const __bf16* p1) {
  v8bf a = *(const v8bf*)p0;
  v8bf b = *(const v8bf*)p1;
  v16bf r;
#pragma unroll
  for (int i = 0; i < 8; ++i) { r[i] = a[i]; r[i + 8] = b[i]; }
  return r;
}

__global__ __launch_bounds__(THREADS, 1)
void ScaledDotProductAttention_fa_bf16_wmma(const float* __restrict__ Q,
                                            const float* __restrict__ K,
                                            const float* __restrict__ V,
                                            float* __restrict__ O) {
  __shared__ __bf16 sQ[QROWS * QSTR];     // Q, bf16, pre-scaled
  __shared__ float  sKf[KTILE * DIM];     // async landing buffer (f32 row-major)
  __shared__ float  sVf[KTILE * DIM];     // async landing buffer (f32 row-major)
  __shared__ __bf16 sKb[KTILE * KSTR];    // K bf16 row-major (B-frags)
  __shared__ __bf16 sVt[DIM * VSTR];      // V bf16 transposed: sVt[d*VSTR + kv]
  __shared__ __bf16 sP[NWAVE * 16 * PSTR];// per-wave P scratch

  const int tid  = threadIdx.x;
  const int lane = tid & 31;
  const int wave = tid >> 5;
  const int n    = lane & 15;    // column / A-row index
  const int hi   = lane >> 4;    // lane half
  const int b    = blockIdx.y;
  const int q0   = blockIdx.x * QROWS;

  // fold softmax scale and log2(e) into Q:  exp(s/16) == exp2(s * 0.0901684...)
  const float qscale = 0.09016844136f;

  const float4* kg = (const float4*)(K + (size_t)b * SEQ * DIM);
  const float4* vg = (const float4*)(V + (size_t)b * SEQ * DIM);

  // ---- stage Q once: f32 -> bf16, pre-scaled ----
  {
    const float4* qg = (const float4*)(Q + ((size_t)b * SEQ + q0) * DIM);
#pragma unroll 4
    for (int i = 0; i < 32; ++i) {
      int idx = i * THREADS + tid;        // float4 index over 128x64
      int row = idx >> 6, c4 = idx & 63;
      float4 v = qg[row * 64 + c4];
      __bf16* d = &sQ[row * QSTR + c4 * 4];
      d[0] = (__bf16)(v.x * qscale); d[1] = (__bf16)(v.y * qscale);
      d[2] = (__bf16)(v.z * qscale); d[3] = (__bf16)(v.w * qscale);
    }
  }

  // ---- prologue: async-DMA tile 0 into the f32 landing buffers ----
#pragma unroll
  for (int i = 0; i < 8; ++i) {
    int idx = i * THREADS + tid;          // float4 index over 32x64
    async_ld_b128(&kg[idx], (unsigned)(uintptr_t)&sKf[idx * 4]);
    async_ld_b128(&vg[idx], (unsigned)(uintptr_t)&sVf[idx * 4]);
  }

  v8f o[16];
#pragma unroll
  for (int t = 0; t < 16; ++t) { v8f z = {}; o[t] = z; }
  float mrow[8], lrow[8];
#pragma unroll
  for (int r = 0; r < 8; ++r) { mrow[r] = -3.0e38f; lrow[r] = 0.0f; }

  __bf16* pS = sP + wave * 16 * PSTR;

  for (int kt = 0; kt < NT; ++kt) {
    wait_async0();          // my async writes for tile kt landed
    __syncthreads();        // everyone's landed; previous compute done

    // ---- LDS convert pass: f32 -> bf16 ----
    {
      // K: row-major, packed v8bf stores
      int row = tid >> 3, cs = (tid & 7) * 32;
      const float* kf = &sKf[row * DIM + cs];
      __bf16*      kb = &sKb[row * KSTR + cs];
#pragma unroll
      for (int j0 = 0; j0 < 32; j0 += 8) {
        v8bf pk;
#pragma unroll
        for (int j = 0; j < 8; ++j) pk[j] = (__bf16)kf[j0 + j];
        *(v8bf*)(kb + j0) = pk;
      }
      // V: transposed; 2 kv-rows x 16 d-cols per thread -> paired b32 stores
      int rp  = tid >> 4;                 // kv row pair 0..15
      int vcs = (tid & 15) * 16;          // d columns
      const float* vf0 = &sVf[(2 * rp + 0) * DIM + vcs];
      const float* vf1 = &sVf[(2 * rp + 1) * DIM + vcs];
#pragma unroll
      for (int j = 0; j < 16; ++j) {
        v2bf pk2 = { (__bf16)vf0[j], (__bf16)vf1[j] };
        *(v2bf*)&sVt[(vcs + j) * VSTR + 2 * rp] = pk2;
      }
    }
    __syncthreads();        // bf16 tiles ready; f32 landing buffers now dead

    // ---- issue async DMA for tile kt+1 (overlaps with compute below) ----
    if (kt + 1 < NT) {
      const float4* kgt = kg + (size_t)(kt + 1) * KTILE * 64;
      const float4* vgt = vg + (size_t)(kt + 1) * KTILE * 64;
#pragma unroll
      for (int i = 0; i < 8; ++i) {
        int idx = i * THREADS + tid;
        async_ld_b128(&kgt[idx], (unsigned)(uintptr_t)&sKf[idx * 4]);
        async_ld_b128(&vgt[idx], (unsigned)(uintptr_t)&sVf[idx * 4]);
      }
    }

    // ---- S = Q K^T : two 16x16 score tiles, contraction over D in 32-chunks
    v8f s0 = {}, s1 = {};
    const __bf16* qbase = sQ + (wave * 16 + n) * QSTR;   // A row = lane&15
#pragma unroll
    for (int dc = 0; dc < 8; ++dc) {
      const int d0 = dc * 32;
      v16bf qa  = ld_frag2(qbase + d0 + 16 * hi, qbase + d0 + 16 * hi + 8);
      v16bf kb0 = ld_frag2(sKb + n * KSTR + d0 + 16 * hi,
                           sKb + n * KSTR + d0 + 16 * hi + 8);
      s0 = __builtin_amdgcn_wmma_f32_16x16x32_bf16(false, qa, false, kb0,
                                                   (short)0, s0, false, false);
      v16bf kb1 = ld_frag2(sKb + (16 + n) * KSTR + d0 + 16 * hi,
                           sKb + (16 + n) * KSTR + d0 + 16 * hi + 8);
      // same A as the previous matrix op -> set matrix-A reuse hint
      s1 = __builtin_amdgcn_wmma_f32_16x16x32_bf16(false, qa, false, kb1,
                                                   (short)0, s1, true, false);
    }

    // ---- online softmax (base-2 domain); rows split across lane halves ----
    float p0[8], p1[8];
    v8f avec;
#pragma unroll
    for (int r = 0; r < 8; ++r) {
      float tmax = fmaxf(s0[r], s1[r]);
#pragma unroll
      for (int off = 1; off < 16; off <<= 1)
        tmax = fmaxf(tmax, __shfl_xor(tmax, off, 32));
      float mnew = fmaxf(mrow[r], tmax);
      float alpha = exp2f(mrow[r] - mnew);
      avec[r]  = alpha;
      mrow[r]  = mnew;
      p0[r] = exp2f(s0[r] - mnew);
      p1[r] = exp2f(s1[r] - mnew);
      float rs = p0[r] + p1[r];
#pragma unroll
      for (int off = 1; off < 16; off <<= 1)
        rs += __shfl_xor(rs, off, 32);
      lrow[r] = lrow[r] * alpha + rs;
    }

    // ---- P tile: C-layout regs -> LDS (bf16, row-major 16x32) ----
#pragma unroll
    for (int r = 0; r < 8; ++r) {
      int m = r + 8 * hi;
      pS[m * PSTR + n]      = (__bf16)p0[r];
      pS[m * PSTR + 16 + n] = (__bf16)p1[r];
    }
    asm volatile("s_wait_dscnt 0" ::: "memory");   // in-wave store->load

    // P as A fragment (16x32)
    const __bf16* pbase = pS + n * PSTR;
    v16bf pa = ld_frag2(pbase + 16 * hi, pbase + 16 * hi + 8);

    // ---- rescale O (packed f32 muls) and accumulate O += P . V ----
    // All 16 PV WMMAs share A == pa; first has no predecessor (reuse=false),
    // the remaining 15 set the matrix-A reuse hint (constant 'true').
    {
      o[0] *= avec;
      const __bf16* vb = sVt + (0 * 16 + n) * VSTR + 16 * hi;
      v16bf vf = ld_frag2(vb, vb + 8);
      o[0] = __builtin_amdgcn_wmma_f32_16x16x32_bf16(false, pa, false, vf,
                                                     (short)0, o[0],
                                                     false, false);
    }
#pragma unroll
    for (int t = 1; t < 16; ++t) {
      o[t] *= avec;                                // -> v_pk_mul_f32 pairs
      const __bf16* vb = sVt + (t * 16 + n) * VSTR + 16 * hi;
      v16bf vf = ld_frag2(vb, vb + 8);
      o[t] = __builtin_amdgcn_wmma_f32_16x16x32_bf16(false, pa, false, vf,
                                                     (short)0, o[t],
                                                     true, false);
    }
  }

  // ---- normalize and write out (fp32) ----
  float inv[8];
#pragma unroll
  for (int r = 0; r < 8; ++r) inv[r] = 1.0f / lrow[r];
  float* outp = O + ((size_t)b * SEQ + q0 + wave * 16) * DIM;
#pragma unroll
  for (int t = 0; t < 16; ++t) {
#pragma unroll
    for (int r = 0; r < 8; ++r) {
      int m = r + 8 * hi;
      outp[(size_t)m * DIM + t * 16 + n] = o[t][r] * inv[r];
    }
  }
}

extern "C" void kernel_launch(void* const* d_in, const int* in_sizes, int n_in,
                              void* d_out, int out_size, void* d_ws, size_t ws_size,
                              hipStream_t stream) {
  (void)in_sizes; (void)n_in; (void)d_ws; (void)ws_size; (void)out_size;
  const float* q = (const float*)d_in[0];
  const float* k = (const float*)d_in[1];
  const float* v = (const float*)d_in[2];
  float* out = (float*)d_out;
  dim3 grid(SEQ / QROWS, BATCH);
  ScaledDotProductAttention_fa_bf16_wmma<<<grid, dim3(THREADS), 0, stream>>>(q, k, v, out);
}
